// KoopmanKernelSeq2Seq_19207093748685
// MI455X (gfx1250) — matrix operand order repaired and checked by reading, not compiled, not run on hardware
//
#include <hip/hip_runtime.h>
#include <math.h>

// ---------------------------------------------------------------------------
// Problem constants (reference: M=512, L=16, O=32, D=64, B=16)
// ---------------------------------------------------------------------------
#define M_    512
#define L_    16
#define O_    32
#define D_    64
#define B_    16
#define ML    8192          // M_ * L_
#define GAMMA_ 0.0078125f   // 1/(2*64)
#define SCALE_ 0.04419417382415922f  // 512^-0.5

typedef __attribute__((ext_vector_type(16))) __bf16 v16bf;
typedef __attribute__((ext_vector_type(8)))  __bf16 v8bf;
typedef __attribute__((ext_vector_type(8)))  float  v8f;

__device__ __forceinline__ unsigned short f2bf(float f) {
  union { float f; unsigned u; } v; v.f = f;
  unsigned r = v.u + 0x7FFFu + ((v.u >> 16) & 1u);   // round-to-nearest-even
  return (unsigned short)(r >> 16);
}

// ---------------------------------------------------------------------------
// koopman fp32 [8192*8192] -> bf16 (so all 32 scan passes stream from L2:
// 128MB bf16 < 192MB L2, while 256MB fp32 would re-stream HBM every step)
// ---------------------------------------------------------------------------
__global__ void cvt_koopman(const float* __restrict__ g,
                            unsigned short* __restrict__ gb, long n4) {
  long i = (long)blockIdx.x * blockDim.x + threadIdx.x;
  long stride = (long)gridDim.x * blockDim.x;
  const float4* g4 = (const float4*)g;
  ushort4* o4 = (ushort4*)gb;
  for (long k = i; k < n4; k += stride) {
    float4 v = g4[k];
    ushort4 r;
    r.x = f2bf(v.x); r.y = f2bf(v.y); r.z = f2bf(v.z); r.w = f2bf(v.w);
    o4[k] = r;
  }
}

// ---------------------------------------------------------------------------
// KX = exp(-g*d2(nys_X, inps)) * scale, written directly as bf16 carry A0:
// A0[b][m*16+l]  (out0 = KX.reshape(B, M*L))
// ---------------------------------------------------------------------------
__global__ void kx_kernel(const float* __restrict__ nysX,
                          const float* __restrict__ inps,
                          unsigned short* __restrict__ A0) {
  __shared__ float xs[D_];
  int m = blockIdx.x;            // 512 blocks
  int t = threadIdx.x;           // 256 threads = 16 b * 16 l
  if (t < D_) xs[t] = nysX[m * D_ + t];
  __syncthreads();
  int b = t >> 4, l = t & 15;
  const float* y = inps + (b * L_ + l) * D_;
  float x2 = 0.f, y2 = 0.f, xy = 0.f;
  #pragma unroll 8
  for (int d = 0; d < D_; ++d) {
    float xv = xs[d], yv = y[d];
    x2 += xv * xv; y2 += yv * yv; xy += xv * yv;
  }
  float d2 = x2 + y2 - 2.f * xy;
  A0[b * ML + m * L_ + l] = f2bf(__expf(-GAMMA_ * d2) * SCALE_);
}

// ---------------------------------------------------------------------------
// KY[i][j] = exp(-g*d2(nys_Y_i, nys_Y_j)) * scale   (f32, [512,512])
// ---------------------------------------------------------------------------
__global__ void ky_kernel(const float* __restrict__ nysY,
                          float* __restrict__ KY) {
  __shared__ float ys[D_];
  int i = blockIdx.x;            // 512 blocks
  int t = threadIdx.x;           // 256 threads
  if (t < D_) ys[t] = nysY[i * D_ + t];
  __syncthreads();
  for (int j = t; j < M_; j += 256) {
    const float* y = nysY + j * D_;
    float x2 = 0.f, y2 = 0.f, xy = 0.f;
    #pragma unroll 8
    for (int d = 0; d < D_; ++d) {
      float a = ys[d], b = y[d];
      x2 += a * a; y2 += b * b; xy += a * b;
    }
    float d2 = x2 + y2 - 2.f * xy;
    KY[i * M_ + j] = __expf(-GAMMA_ * d2) * SCALE_;
  }
}

// ---------------------------------------------------------------------------
// Wt[m][a] = sum_i nys_Y[i][a] * KY[i][m]   ([512,64])
// (collapses the last two einsums into one [64,512] projection)
// ---------------------------------------------------------------------------
__global__ void wt_kernel(const float* __restrict__ nysY,
                          const float* __restrict__ KY,
                          float* __restrict__ Wt) {
  int mcol = blockIdx.x;   // 512
  int a = threadIdx.x;     // 64
  float acc = 0.f;
  for (int i = 0; i < M_; ++i)
    acc += nysY[i * D_ + a] * KY[i * M_ + mcol];  // KY read is scalar-broadcast
  Wt[mcol * D_ + a] = acc;
}

// ---------------------------------------------------------------------------
// One scan step: Anext[16, 8192] = A[16, 8192] @ G^T   (G = koopman bf16)
// v_wmma_f32_16x16x32_bf16; batch B=16 exactly fills the M dim of one tile.
// 512 waves (128 blocks x 4 waves), each owns ONE 16-col N-tile across all
// of K=8192, with manual register double-buffering: iteration k+1's operands
// are loaded before iteration k's WMMAs consume theirs, so the scheduler can
// overlap a full iteration of L2 latency (partial s_wait_loadcnt) instead of
// the round-1 pattern of s_wait_loadcnt 0x0 before every WMMA.
// Operand layouts used here (A and B share the same contiguous k order):
//   A (16x32 bf16): lane m=l&15, hi=l>>4; halves 0-7 = K[k0+16*hi ..),
//                   halves 8-15 = K[k0+16*hi+8 ..)   -> two 16B loads
//   B (32x16 bf16): lane n=l&15, hi=l>>4; halves 0-15 = K[k0+16*hi ..)
//                   -> one contiguous 32B load per lane
//   D (16x16 f32):  VGPR p, row = p + 8*hi, col = n0 + (l&15)
// ---------------------------------------------------------------------------
__global__ __launch_bounds__(128)
void scan_step(const unsigned short* __restrict__ A,
               const unsigned short* __restrict__ G,
               unsigned short* __restrict__ Anext,
               float* __restrict__ outs_o) {
  int wave = blockIdx.x * (blockDim.x >> 5) + (threadIdx.x >> 5);
  int lane = threadIdx.x & 31;
  int hi = lane >> 4;
  int lm = lane & 15;
  int n0 = wave * 16;            // this wave owns columns [n0, n0+16)

  const unsigned short* aRow = A + lm * ML + hi * 16;
  const unsigned short* bRow = G + (long)(n0 + lm) * ML + hi * 16;

  v8f acc = {};

  // prologue: load k0 = 0 operands
  union { v16bf v; v8bf h[2]; } a_c, a_n;
  a_c.h[0] = *(const v8bf*)(aRow);
  a_c.h[1] = *(const v8bf*)(aRow + 8);
  v16bf b_c = *(const v16bf*)(bRow);
  v16bf b_n;

  #pragma unroll 2
  for (int k0 = 0; k0 < ML - 32; k0 += 32) {
    // issue next iteration's loads first (double buffer in registers)
    a_n.h[0] = *(const v8bf*)(aRow + k0 + 32);
    a_n.h[1] = *(const v8bf*)(aRow + k0 + 40);
    b_n = *(const v16bf*)(bRow + k0 + 32);

    acc = __builtin_amdgcn_wmma_f32_16x16x32_bf16(
        false, a_c.v, false, b_c, (short)0, acc, false, false);

    a_c.v = a_n.v;
    b_c = b_n;
  }
  // epilogue: last k-tile
  acc = __builtin_amdgcn_wmma_f32_16x16x32_bf16(
      false, a_c.v, false, b_c, (short)0, acc, false, false);

  int col = n0 + lm;
  #pragma unroll
  for (int p = 0; p < 8; ++p) {
    int row = p + hi * 8;                      // batch row b
    float v = acc[p];
    outs_o[row * ML + col] = v;                // f32 history for projection
    Anext[row * ML + col] = f2bf(v);           // bf16 carry for next step
  }
}

// ---------------------------------------------------------------------------
// out[b,l,o,a] = sum_m Wt[m][a] * outs[o][b][m*16+l]
// One block per (b,l); all 32 o-slices (512 floats each) staged in 64KB LDS,
// then Wt streamed coalesced with LDS broadcasts.
// ---------------------------------------------------------------------------
__global__ __launch_bounds__(64)
void project_kernel(const float* __restrict__ outs,
                    const float* __restrict__ Wt,
                    float* __restrict__ out) {
  __shared__ float S[O_][M_];                  // 32*512*4 = 64KB
  int b = blockIdx.x >> 4;
  int l = blockIdx.x & 15;
  int a = threadIdx.x;                         // 64 threads

  for (int o = 0; o < O_; ++o) {
    const float* src = outs + ((long)o * B_ + b) * ML + l;   // element m: src[m*16]
    for (int m = a; m < M_; m += 64) S[o][m] = src[m * L_];
  }
  __syncthreads();

  float acc[O_];
  #pragma unroll
  for (int o = 0; o < O_; ++o) acc[o] = 0.f;

  for (int m = 0; m < M_; ++m) {
    float w = Wt[m * D_ + a];                  // coalesced
    #pragma unroll
    for (int o = 0; o < O_; ++o) acc[o] += w * S[o][m];  // LDS broadcast
  }

  long base = (((long)b * L_ + l) * O_) * D_ + a;
  #pragma unroll
  for (int o = 0; o < O_; ++o) out[base + (long)o * D_] = acc[o];
}

// ---------------------------------------------------------------------------
extern "C" void kernel_launch(void* const* d_in, const int* in_sizes, int n_in,
                              void* d_out, int out_size, void* d_ws, size_t ws_size,
                              hipStream_t stream) {
  const float* inps = (const float*)d_in[0];   // [16,16,64]
  const float* nysX = (const float*)d_in[1];   // [512,64]
  const float* nysY = (const float*)d_in[2];   // [512,64]
  const float* koop = (const float*)d_in[3];   // [8192,8192]
  float* out = (float*)d_out;                  // [16,16,32,64] f32

  char* ws = (char*)d_ws;
  size_t off = 0;
  auto wsalloc = [&](size_t bytes) -> void* {
    void* p = ws + off;
    off += (bytes + 255) & ~(size_t)255;
    return p;
  };
  unsigned short* Gb   = (unsigned short*)wsalloc((size_t)ML * ML * 2);   // 128MB
  unsigned short* Abuf0= (unsigned short*)wsalloc((size_t)B_ * ML * 2);   // 256KB
  unsigned short* Abuf1= (unsigned short*)wsalloc((size_t)B_ * ML * 2);   // 256KB
  float* outsBuf       = (float*)wsalloc((size_t)O_ * B_ * ML * 4);       // 16MB
  float* KY            = (float*)wsalloc((size_t)M_ * M_ * 4);            // 1MB
  float* Wt            = (float*)wsalloc((size_t)M_ * D_ * 4);            // 128KB
  (void)ws_size; (void)in_sizes; (void)n_in; (void)out_size;

  // Setup (independent; all on one stream)
  cvt_koopman<<<8192, 256, 0, stream>>>(koop, Gb, (long)ML * ML / 4);
  kx_kernel<<<M_, 256, 0, stream>>>(nysX, inps, Abuf0);
  ky_kernel<<<M_, 256, 0, stream>>>(nysY, KY);
  wt_kernel<<<M_, D_, 0, stream>>>(nysY, KY, Wt);

  // Sequential Koopman scan: 32 dependent bf16 WMMA GEMM steps
  unsigned short* Acur = Abuf0;
  unsigned short* Anxt = Abuf1;
  for (int o = 0; o < O_; ++o) {
    scan_step<<<128, 128, 0, stream>>>(Acur, Gb, Anxt,
                                       outsBuf + (size_t)o * B_ * ML);
    unsigned short* t = Acur; Acur = Anxt; Anxt = t;
  }

  // Final fused projection (nys_Y^T * KY) applied to all step outputs
  project_kernel<<<B_ * L_, D_, 0, stream>>>(outsBuf, Wt, out);
}
